// BiMamba_Block_82635170775462
// MI455X (gfx1250) — matrix operand order repaired and checked by reading, not compiled, not run on hardware
//
#include <hip/hip_runtime.h>
#include <hip/hip_bf16.h>
#include <math.h>

// Problem constants (from reference)
#define BB  8
#define LL  1024
#define DMC 512
#define DIC 1024
#define NN  16
#define RR  32
#define KCV 4
#define MM  (BB * LL)   // 8192 rows

typedef __attribute__((ext_vector_type(16))) __bf16 v16bf;
typedef __attribute__((ext_vector_type(8)))  __bf16 v8bf;
typedef __attribute__((ext_vector_type(8)))  float  v8f;

__device__ __forceinline__ float sigmoid_f(float x) { return 1.0f / (1.0f + __expf(-x)); }

// Load 16 contiguous bf16 (32B): WMMA B-fragment
__device__ __forceinline__ v16bf ld_bf16x16(const __bf16* p) {
    v8bf lo = *(const v8bf*)(p);
    v8bf hi = *(const v8bf*)(p + 8);
    return __builtin_shufflevector(lo, hi, 0,1,2,3,4,5,6,7,8,9,10,11,12,13,14,15);
}
// Load WMMA A-fragment: 8 bf16 at p, 8 bf16 at p+16 (ISA 7.12.2 16-bit A 16x32)
__device__ __forceinline__ v16bf ld_afrag(const __bf16* p) {
    v8bf lo = *(const v8bf*)(p);
    v8bf hi = *(const v8bf*)(p + 16);
    return __builtin_shufflevector(lo, hi, 0,1,2,3,4,5,6,7,8,9,10,11,12,13,14,15);
}

#define WMMA_BF16(A_, B_, C_) \
    __builtin_amdgcn_wmma_f32_16x16x32_bf16(false, (A_), false, (B_), (short)0, (C_), false, false)

// ---------------------------------------------------------------------------
// Register-blocked WMMA GEMM: C[M x Nc] = A[M x K] * B[K x Nc]
//   A  : bf16, row-major, leading dim lda
//   Bt : bf16, pre-transposed [Nc][K]
//   C  : f32 out; Cbf optional bf16 mirror (same ldc)
// One wave computes a 32x64 tile (2x4 WMMA register block): 8 v_wmma per
// k-step from 12 x 16B loads (24 B/WMMA). Work units (m-tile-pair, n-quad)
// are flattened across waves for load balance on narrow Nc.
// flipL!=0 remaps A rows (b,t) -> (b, L-1-t) for the backward direction.
// ---------------------------------------------------------------------------
__global__ __launch_bounds__(256) void wmma_gemm_bf16_kernel(
    const __bf16* __restrict__ A, int lda,
    const __bf16* __restrict__ Bt,
    float* __restrict__ C, int ldc,
    __bf16* __restrict__ Cbf,
    int M, int Nc, int K, int flipL, int Lseq)
{
    const int lane = threadIdx.x & 31;
    const int wave = threadIdx.x >> 5;
    const int ntiles = Nc >> 4;
    const int nquads = (ntiles + 3) >> 2;        // 64-column quads
    const int munits = M >> 5;                   // 32-row units
    const int unit = blockIdx.x * 8 + wave;
    if (unit >= munits * nquads) return;
    const int mq = unit / nquads;
    const int q  = unit - mq * nquads;

    const int m0 = mq << 5;
    const int n0 = q << 6;

    // A fragment rows (lanes 0-15 K=0..15, lanes 16-31 K=16..31)
    int arow0 = m0 + (lane & 15);
    int arow1 = arow0 + 16;
    if (flipL) {
        int b0 = arow0 / Lseq, t0 = arow0 - b0 * Lseq;
        int b1 = arow1 / Lseq, t1 = arow1 - b1 * Lseq;
        arow0 = b0 * Lseq + (Lseq - 1 - t0);
        arow1 = b1 * Lseq + (Lseq - 1 - t1);
    }
    const int ka = (lane < 16) ? 0 : 16;
    const int kb = (lane < 16) ? 0 : 16;
    const int nc0 = n0 + (lane & 15);

    const __bf16* aRow0 = A  + (size_t)arow0 * lda + ka;
    const __bf16* aRow1 = A  + (size_t)arow1 * lda + ka;
    const __bf16* bCol  = Bt + (size_t)nc0  * K  + kb;

    v8f acc00 = {}, acc01 = {}, acc02 = {}, acc03 = {};
    v8f acc10 = {}, acc11 = {}, acc12 = {}, acc13 = {};

    for (int k0 = 0; k0 < K; k0 += 32) {
        __builtin_prefetch(aRow0 + k0 + 32, 0, 1);   // global_prefetch_b8
        __builtin_prefetch(aRow1 + k0 + 32, 0, 1);

        v16bf af0 = ld_bf16x16(aRow0 + k0);
        v16bf af1 = ld_bf16x16(aRow1 + k0);

        const __bf16* bp = bCol + k0;
        v16bf bf0 = ld_bf16x16(bp);
        v16bf bf1 = ld_bf16x16(bp + (size_t)16 * K);
        v16bf bf2 = ld_bf16x16(bp + (size_t)32 * K);
        v16bf bf3 = ld_bf16x16(bp + (size_t)48 * K);

        acc00 = WMMA_BF16(af0, bf0, acc00);
        acc01 = WMMA_BF16(af0, bf1, acc01);
        acc02 = WMMA_BF16(af0, bf2, acc02);
        acc03 = WMMA_BF16(af0, bf3, acc03);
        acc10 = WMMA_BF16(af1, bf0, acc10);
        acc11 = WMMA_BF16(af1, bf1, acc11);
        acc12 = WMMA_BF16(af1, bf2, acc12);
        acc13 = WMMA_BF16(af1, bf3, acc13);
    }

    // C/D layout: VGPR r -> row base + r + 8*(lane>=16), col = n + (lane&15)
    const int crow0 = m0 + ((lane >> 4) << 3);
    v8f accs[2][4] = { { acc00, acc01, acc02, acc03 },
                       { acc10, acc11, acc12, acc13 } };
    #pragma unroll
    for (int mi = 0; mi < 2; ++mi) {
        const int crow = crow0 + 16 * mi;
        #pragma unroll
        for (int i = 0; i < 4; ++i) {
            const int ncol = nc0 + 16 * i;
            #pragma unroll
            for (int r = 0; r < 8; ++r) {
                const size_t off = (size_t)(crow + r) * ldc + ncol;
                C[off] = accs[mi][i][r];
                if (Cbf) Cbf[off] = (__bf16)accs[mi][i][r];
            }
        }
    }
}

// ---------------------------------------------------------------------------
// Weight prep: Bt[n*K + k] = (bf16) W[k*Nc + n]   (transpose + cast)
// ---------------------------------------------------------------------------
__global__ __launch_bounds__(256) void weight_transpose_cast_kernel(
    const float* __restrict__ W, __bf16* __restrict__ Bt, int K, int Nc)
{
    int idx = blockIdx.x * 256 + threadIdx.x;
    if (idx >= K * Nc) return;
    int k = idx / Nc, n = idx - k * Nc;
    Bt[(size_t)n * K + k] = (__bf16)W[idx];
}

// Plain f32 -> bf16 cast
__global__ __launch_bounds__(256) void cast_bf16_kernel(
    const float* __restrict__ src, __bf16* __restrict__ dst, int n)
{
    int idx = blockIdx.x * 256 + threadIdx.x;
    if (idx < n) dst[idx] = (__bf16)src[idx];
}

// ---------------------------------------------------------------------------
// Causal depthwise conv (K=4) + SiLU; writes f32 (scan input) + bf16 (GEMM A)
// ---------------------------------------------------------------------------
__global__ __launch_bounds__(256) void conv_silu_kernel(
    const float* __restrict__ xz, const float* __restrict__ convw,
    const float* __restrict__ convb, float* __restrict__ xc,
    __bf16* __restrict__ xcbf)
{
    int idx = blockIdx.x * 256 + threadIdx.x;    // over MM*DIC
    int m = idx >> 10;                           // DIC = 1024
    int d = idx & (DIC - 1);
    int b = m >> 10;                             // LL = 1024
    int t = m & (LL - 1);
    float acc = convb[d];
    #pragma unroll
    for (int k = 0; k < KCV; ++k) {
        int tt = t - (KCV - 1) + k;
        if (tt >= 0)
            acc += xz[((size_t)(b * LL + tt)) * (2 * DIC) + d] * convw[d * KCV + k];
    }
    float s = acc * sigmoid_f(acc);
    xc[idx] = s;
    xcbf[idx] = (__bf16)s;
}

// dt = softplus(dtmm + bdt), in-place
__global__ __launch_bounds__(256) void softplus_kernel(
    float* __restrict__ dtb, const float* __restrict__ bdt)
{
    int idx = blockIdx.x * 256 + threadIdx.x;
    int d = idx & (DIC - 1);
    float v = dtb[idx] + bdt[d];
    dtb[idx] = (v > 20.0f) ? v : log1pf(__expf(v));
}

// ---------------------------------------------------------------------------
// Selective scan: one thread per (b,d) channel, NN=16 states in VGPRs.
// B_t/C_t rows are block-uniform -> compiler scalarizes to SMEM loads.
// ---------------------------------------------------------------------------
__global__ __launch_bounds__(256) void scan_kernel(
    const float* __restrict__ dt, const float* __restrict__ dbc,
    const float* __restrict__ xc, const float* __restrict__ A_log,
    float* __restrict__ ys)
{
    const int chunks = DIC / 256;                // 4
    const int b = blockIdx.x / chunks;
    const int d = (blockIdx.x % chunks) * 256 + threadIdx.x;

    float Ar[NN];
    #pragma unroll
    for (int n = 0; n < NN; ++n) Ar[n] = -__expf(A_log[(size_t)d * NN + n]);

    float h[NN];
    #pragma unroll
    for (int n = 0; n < NN; ++n) h[n] = 0.0f;

    for (int t = 0; t < LL; ++t) {
        const int m = b * LL + t;
        const float dtv = dt[(size_t)m * DIC + d];
        const float u   = xc[(size_t)m * DIC + d];
        const float du  = dtv * u;
        const float* bc = dbc + (size_t)m * (RR + 2 * NN);
        float y = 0.0f;
        #pragma unroll
        for (int n = 0; n < NN; ++n) {
            float dA = __expf(dtv * Ar[n]);
            h[n] = dA * h[n] + du * bc[RR + n];
            y += h[n] * bc[RR + NN + n];
        }
        ys[(size_t)m * DIC + d] = y;
    }
}

// Gate: ysbf = bf16( (ys + xc*D) * silu(z) ),  z = xz[:, DI + d]
__global__ __launch_bounds__(256) void gate_kernel(
    const float* __restrict__ ys, const float* __restrict__ xc,
    const float* __restrict__ xz, const float* __restrict__ Dp,
    __bf16* __restrict__ ysbf)
{
    int idx = blockIdx.x * 256 + threadIdx.x;
    int m = idx >> 10;
    int d = idx & (DIC - 1);
    float z = xz[(size_t)m * (2 * DIC) + DIC + d];
    ysbf[idx] = (__bf16)((ys[idx] + xc[idx] * Dp[d]) * (z * sigmoid_f(z)));
}

// ---------------------------------------------------------------------------
// Residual + LayerNorm: out = LN(x + yf + flip(yb)) * g + b   (block per row)
// ---------------------------------------------------------------------------
__global__ __launch_bounds__(256) void ln_kernel(
    const float* __restrict__ x, const float* __restrict__ yf,
    const float* __restrict__ yb, const float* __restrict__ g,
    const float* __restrict__ bb, float* __restrict__ out)
{
    const int m = blockIdx.x;
    const int b = m >> 10, t = m & (LL - 1);
    const int mf = b * LL + (LL - 1 - t);
    const int tid = threadIdx.x;

    float v0 = x[(size_t)m * DMC + tid]       + yf[(size_t)m * DMC + tid]
             + yb[(size_t)mf * DMC + tid];
    float v1 = x[(size_t)m * DMC + tid + 256] + yf[(size_t)m * DMC + tid + 256]
             + yb[(size_t)mf * DMC + tid + 256];

    __shared__ float s1[256], s2[256];
    s1[tid] = v0 + v1;
    s2[tid] = v0 * v0 + v1 * v1;
    __syncthreads();
    for (int off = 128; off > 0; off >>= 1) {
        if (tid < off) { s1[tid] += s1[tid + off]; s2[tid] += s2[tid + off]; }
        __syncthreads();
    }
    const float mean = s1[0] * (1.0f / DMC);
    const float var  = s2[0] * (1.0f / DMC) - mean * mean;
    const float rstd = rsqrtf(var + 1e-5f);

    out[(size_t)m * DMC + tid]       = (v0 - mean) * rstd * g[tid]       + bb[tid];
    out[(size_t)m * DMC + tid + 256] = (v1 - mean) * rstd * g[tid + 256] + bb[tid + 256];
}

// ---------------------------------------------------------------------------
extern "C" void kernel_launch(void* const* d_in, const int* in_sizes, int n_in,
                              void* d_out, int out_size, void* d_ws, size_t ws_size,
                              hipStream_t stream)
{
    const float* x    = (const float*)d_in[0];
    const float* ln_g = (const float*)d_in[19];
    const float* ln_b = (const float*)d_in[20];
    float* out = (float*)d_out;

    // ---- workspace layout: f32 region then bf16 region ----
    float* ws  = (float*)d_ws;
    float* xz  = ws;                                   // MM*2048
    float* xc  = xz  + (size_t)MM * 2 * DIC;           // MM*1024
    float* dbc = xc  + (size_t)MM * DIC;               // MM*64
    float* dtb = dbc + (size_t)MM * (RR + 2 * NN);     // MM*1024
    float* ys  = dtb + (size_t)MM * DIC;               // MM*1024
    float* ydf = ys  + (size_t)MM * DIC;               // MM*512
    float* ydb = ydf + (size_t)MM * DMC;               // MM*512

    __bf16* bfbase = (__bf16*)(ydb + (size_t)MM * DMC);
    __bf16* xbf    = bfbase;                            // MM*512
    __bf16* xcbf   = xbf   + (size_t)MM * DMC;          // MM*1024
    __bf16* dbcbf  = xcbf  + (size_t)MM * DIC;          // MM*64
    __bf16* ysbf   = dbcbf + (size_t)MM * (RR + 2*NN);  // MM*1024
    __bf16* WinT   = ysbf  + (size_t)MM * DIC;          // 2048*512
    __bf16* WxT    = WinT  + (size_t)(2 * DIC) * DMC;   // 64*1024
    __bf16* WdtT   = WxT   + (size_t)(RR + 2*NN) * DIC; // 1024*32
    __bf16* WoutT  = WdtT  + (size_t)DIC * RR;          // 512*1024

    auto gemm_grid = [](int M, int Nc) -> int {
        int ntiles = Nc >> 4;
        int nquads = (ntiles + 3) >> 2;
        int units  = (M >> 5) * nquads;
        return (units + 7) / 8;
    };
    const int ew_blocks = (MM * DIC) / 256;             // 32768

    // x -> bf16 once (row flip for bwd handled inside GEMM addressing)
    cast_bf16_kernel<<<(MM * DMC) / 256, 256, 0, stream>>>(x, xbf, MM * DMC);

    for (int dir = 0; dir < 2; ++dir) {
        const float* Win   = (const float*)d_in[1 + 9 * dir];
        const float* convw = (const float*)d_in[2 + 9 * dir];
        const float* convb = (const float*)d_in[3 + 9 * dir];
        const float* Wx    = (const float*)d_in[4 + 9 * dir];
        const float* Wdt   = (const float*)d_in[5 + 9 * dir];
        const float* bdt   = (const float*)d_in[6 + 9 * dir];
        const float* Alog  = (const float*)d_in[7 + 9 * dir];
        const float* Dp    = (const float*)d_in[8 + 9 * dir];
        const float* Wout  = (const float*)d_in[9 + 9 * dir];
        float* ydir = dir ? ydb : ydf;
        const int flip = dir;

        // weight prep: transpose+cast to bf16 [N][K]
        weight_transpose_cast_kernel<<<(DMC * 2 * DIC) / 256, 256, 0, stream>>>(Win,  WinT,  DMC, 2 * DIC);
        weight_transpose_cast_kernel<<<(DIC * (RR + 2*NN)) / 256, 256, 0, stream>>>(Wx, WxT, DIC, RR + 2 * NN);
        weight_transpose_cast_kernel<<<(RR * DIC) / 256, 256, 0, stream>>>(Wdt,  WdtT,  RR,  DIC);
        weight_transpose_cast_kernel<<<(DIC * DMC) / 256, 256, 0, stream>>>(Wout, WoutT, DIC, DMC);

        // 1) xz = (flip? rev(x) : x) @ Win            [8192 x 512 x 2048]
        wmma_gemm_bf16_kernel<<<gemm_grid(MM, 2 * DIC), 256, 0, stream>>>(
            xbf, DMC, WinT, xz, 2 * DIC, (__bf16*)nullptr, MM, 2 * DIC, DMC, flip, LL);
        // 2) xc = silu(conv(xi))  (f32 + bf16)
        conv_silu_kernel<<<ew_blocks, 256, 0, stream>>>(xz, convw, convb, xc, xcbf);
        // 3) dbc = xc @ Wx                            [8192 x 1024 x 64], dual-write
        wmma_gemm_bf16_kernel<<<gemm_grid(MM, RR + 2 * NN), 256, 0, stream>>>(
            xcbf, DIC, WxT, dbc, RR + 2 * NN, dbcbf, MM, RR + 2 * NN, DIC, 0, LL);
        // 4) dt = dbc[:, :32] @ Wdt                   [8192 x 32 x 1024]
        wmma_gemm_bf16_kernel<<<gemm_grid(MM, DIC), 256, 0, stream>>>(
            dbcbf, RR + 2 * NN, WdtT, dtb, DIC, (__bf16*)nullptr, MM, DIC, RR, 0, LL);
        softplus_kernel<<<ew_blocks, 256, 0, stream>>>(dtb, bdt);
        // 5) selective scan
        scan_kernel<<<BB * (DIC / 256), 256, 0, stream>>>(dtb, dbc, xc, Alog, ys);
        // 6) gate -> bf16 GEMM input
        gate_kernel<<<ew_blocks, 256, 0, stream>>>(ys, xc, xz, Dp, ysbf);
        // 7) ydir = gated @ Wout                      [8192 x 1024 x 512]
        wmma_gemm_bf16_kernel<<<gemm_grid(MM, DMC), 256, 0, stream>>>(
            ysbf, DIC, WoutT, ydir, DMC, (__bf16*)nullptr, MM, DMC, DIC, 0, LL);
    }

    // 8) out = LN(x + yf + flip(yb)) * g + b
    ln_kernel<<<MM, 256, 0, stream>>>(x, ydf, ydb, ln_g, ln_b, out);
}
